// BasicBlock2_27934467293307
// MI455X (gfx1250) — 8-bit matrix operand order repaired and checked by reading, compiled, not run on hardware
//
#include <hip/hip_runtime.h>
#include <hip/hip_bf16.h>
#include <stdint.h>

typedef __attribute__((ext_vector_type(8))) int v8i;

#define NIMG 32
#define CH   64
#define HH   112
#define WW   112
#define KSZ  576
#define HWSZ (HH*WW)
#define TOT  (NIMG*CH*HWSZ)
#define BN_EPS 1e-5f

__global__ __launch_bounds__(256)
void k_binarize_x(const float* __restrict__ x, uint32_t* __restrict__ xb4, int n4) {
  int i = blockIdx.x * 256 + threadIdx.x;
  if (i >= n4) return;
  float4 v = reinterpret_cast<const float4*>(x)[i];
  uint32_t p = 0;
  p |=  (uint32_t)(uint8_t)(signed char)(v.x >= 0.f ? 1 : -1);
  p |= ((uint32_t)(uint8_t)(signed char)(v.y >= 0.f ? 1 : -1)) << 8;
  p |= ((uint32_t)(uint8_t)(signed char)(v.z >= 0.f ? 1 : -1)) << 16;
  p |= ((uint32_t)(uint8_t)(signed char)(v.w >= 0.f ? 1 : -1)) << 24;
  xb4[i] = p;
}

__global__ __launch_bounds__(128)
void k_prep_afrag(const float* __restrict__ Wt, uint32_t* __restrict__ afrag) {
  int t = blockIdx.x * 128 + threadIdx.x;
  if (t >= 4 * 9 * 32) return;
  int mtile  = t / 288;
  int rem    = t % 288;
  int s      = rem / 32;
  int lane   = rem % 32;
  int laneHi = lane >> 4;
  int m      = mtile * 16 + (lane & 15);
  #pragma unroll
  for (int g = 0; g < 8; ++g) {
    int kb = s * 64 + ((g >> 1) * 16) + ((g & 1) * 4) + laneHi * 8;
    uint32_t p = 0;
    #pragma unroll
    for (int q = 0; q < 4; ++q) {
      int k = kb + q;
      signed char b = (Wt[m * KSZ + k] >= 0.f) ? 1 : -1;
      p |= ((uint32_t)(uint8_t)b) << (8 * q);
    }
    afrag[t * 8 + g] = p;
  }
}

__global__ void k_alpha(const float* __restrict__ Wt, float* __restrict__ alpha) {
  int o = threadIdx.x;
  if (o >= CH) return;
  float s = 0.f;
  for (int j = 0; j < KSZ; ++j) s += fabsf(Wt[o * KSZ + j]);
  alpha[o] = s / (float)KSZ;
}

__global__ __launch_bounds__(128)
void k_conv_wmma(const signed char* __restrict__ xb,
                 const v8i* __restrict__ afrag,
                 const float* __restrict__ alpha,
                 short* __restrict__ convs,
                 float* __restrict__ gsum, float* __restrict__ gsq) {
  __shared__ signed char xt[CH][3][20];
  __shared__ __align__(32) uint32_t blds[9 * 32 * 8];
  __shared__ float s_sum[CH];
  __shared__ float s_sq[CH];

  const int tid = threadIdx.x;
  const int w0  = blockIdx.x * 16;
  const int h   = blockIdx.y;
  const int n   = blockIdx.z;

  if (tid < CH) { s_sum[tid] = 0.f; s_sq[tid] = 0.f; }

  __builtin_prefetch(&afrag[(tid >> 5) * 9 * 32 + (tid & 31)], 0, 0);

  for (int t = tid; t < CH * 3 * 18; t += 128) {
    int ci  = t / 54;
    int r   = t % 54;
    int kh  = r / 18;
    int col = r % 18;
    int ih  = h + kh - 1;
    int iw  = w0 + col - 1;
    signed char v = 0;
    if ((unsigned)ih < HH && (unsigned)iw < WW)
      v = xb[((size_t)(n * CH + ci) * HH + ih) * WW + iw];
    xt[ci][kh][col] = v;
  }
  __syncthreads();

  for (int t = tid; t < 9 * 32 * 8; t += 128) {
    int s    = t >> 8;
    int r    = t & 255;
    int lane = r >> 3;
    int g    = r & 7;
    int laneHi = lane >> 4;
    int ncol   = lane & 15;
    int kb = s * 64 + ((g >> 1) * 16) + ((g & 1) * 4) + laneHi * 8;
    uint32_t p = 0;
    #pragma unroll
    for (int q = 0; q < 4; ++q) {
      int k  = kb + q;
      int ci = k / 9;
      int rr = k - ci * 9;
      int kh = rr / 3;
      int kw = rr - kh * 3;
      p |= ((uint32_t)(uint8_t)xt[ci][kh][ncol + kw]) << (8 * q);
    }
    blds[(s * 32 + lane) * 8 + g] = p;
  }
  __syncthreads();

  const int lane   = tid & 31;
  const int mtile  = tid >> 5;
  const int laneHi = lane >> 4;
  const int ncol   = lane & 15;

  v8i acc = {};
  #pragma unroll
  for (int s = 0; s < 9; ++s) {
    v8i a = afrag[(mtile * 9 + s) * 32 + lane];
    v8i b = *reinterpret_cast<const v8i*>(&blds[(s * 32 + lane) * 8]);
    acc = __builtin_amdgcn_wmma_i32_16x16x64_iu8(true, a, true, b, acc,
                                                 false, false);
  }

  #pragma unroll
  for (int v = 0; v < 8; ++v) {
    int   cout = mtile * 16 + v + laneHi * 8;
    int   ival = acc[v];
    float val  = (float)ival * alpha[cout];
    convs[((size_t)(n * CH + cout) * HH + h) * WW + w0 + ncol] = (short)ival;
    atomicAdd(&s_sum[cout], val);
    atomicAdd(&s_sq[cout],  val * val);
  }
  __syncthreads();
  if (tid < CH) {
    atomicAdd(&gsum[tid], s_sum[tid]);
    atomicAdd(&gsq[tid],  s_sq[tid]);
  }
}

__global__ void k_stats(const float* __restrict__ gsum, const float* __restrict__ gsq,
                        const float* __restrict__ alpha,
                        const float* __restrict__ gamma, const float* __restrict__ beta,
                        float* __restrict__ scale, float* __restrict__ shift) {
  int c = threadIdx.x;
  if (c >= CH) return;
  const float cnt = (float)(NIMG * HWSZ);
  float mu  = gsum[c] / cnt;
  float var = gsq[c] / cnt - mu * mu;
  float rs  = rsqrtf(var + BN_EPS);
  float g   = gamma[c];
  scale[c] = alpha[c] * rs * g;
  shift[c] = beta[c] - mu * rs * g;
}

__global__ __launch_bounds__(256)
void k_apply(const short* __restrict__ convs, const float* __restrict__ x,
             const float* __restrict__ scale, const float* __restrict__ shift,
             float* __restrict__ out, int n4) {
  int i = blockIdx.x * 256 + threadIdx.x;
  if (i >= n4) return;
  int c = (i / (HWSZ / 4)) % CH;
  float sc = scale[c], sh = shift[c];
  short4 cv = reinterpret_cast<const short4*>(convs)[i];
  float4 xv = reinterpret_cast<const float4*>(x)[i];
  float4 o;
  o.x = fminf(fmaxf((float)cv.x * sc + sh, -1.f), 1.f) + xv.x;
  o.y = fminf(fmaxf((float)cv.y * sc + sh, -1.f), 1.f) + xv.y;
  o.z = fminf(fmaxf((float)cv.z * sc + sh, -1.f), 1.f) + xv.z;
  o.w = fminf(fmaxf((float)cv.w * sc + sh, -1.f), 1.f) + xv.w;
  reinterpret_cast<float4*>(out)[i] = o;
}

extern "C" void kernel_launch(void* const* d_in, const int* in_sizes, int n_in,
                              void* d_out, int out_size, void* d_ws, size_t ws_size,
                              hipStream_t stream) {
  const float* x     = (const float*)d_in[0];
  const float* Wt    = (const float*)d_in[1];
  const float* gamma = (const float*)d_in[2];
  const float* beta  = (const float*)d_in[3];
  float* out = (float*)d_out;
  char*  ws  = (char*)d_ws;

  size_t off = 0;
  short*       convs = (short*)(ws + off);       off += (size_t)TOT * 2;
  signed char* xb    = (signed char*)(ws + off); off += (size_t)TOT;
  uint32_t*    afrag = (uint32_t*)(ws + off);    off += 4 * 9 * 32 * 32;
  float*       alpha = (float*)(ws + off);       off += 256;
  float*       gsum  = (float*)(ws + off);       off += 256;
  float*       gsq   = (float*)(ws + off);       off += 256;
  float*       scale = (float*)(ws + off);       off += 256;
  float*       shift = (float*)(ws + off);       off += 256;
  (void)off; (void)ws_size; (void)in_sizes; (void)n_in; (void)out_size;

  hipMemsetAsync(gsum, 0, 512, stream);

  const int n4 = TOT / 4;
  k_binarize_x<<<n4 / 256, 256, 0, stream>>>(x, (uint32_t*)xb, n4);
  k_prep_afrag<<<9, 128, 0, stream>>>(Wt, afrag);
  k_alpha<<<1, 64, 0, stream>>>(Wt, alpha);

  dim3 cg(WW / 16, HH, NIMG);
  k_conv_wmma<<<cg, 128, 0, stream>>>(xb, (const v8i*)afrag, alpha,
                                      convs, gsum, gsq);

  k_stats<<<1, 64, 0, stream>>>(gsum, gsq, alpha, gamma, beta, scale, shift);
  k_apply<<<n4 / 256, 256, 0, stream>>>(convs, x, scale, shift, out, n4);
}
